// MultiHeadAttention_13151189860665
// MI455X (gfx1250) — matrix operand order repaired and checked by reading, not compiled, not run on hardware
//
#include <hip/hip_runtime.h>
#include <hip/hip_bf16.h>

typedef __attribute__((ext_vector_type(16))) _Float16     v16h;
typedef __attribute__((ext_vector_type(8)))  float        v8f;
typedef __attribute__((ext_vector_type(4)))  unsigned int u32x4;
typedef __attribute__((ext_vector_type(8)))  unsigned int u32x8;

#define WMMA_F16(a, b, c) \
    __builtin_amdgcn_wmma_f32_16x16x32_f16(false, (a), false, (b), (short)0, (c), false, false)

// LDS byte offset of a __shared__ object (ptrtoint of addrspace(3) pointer)
#define LDS_OFFSET(p) ((unsigned)(unsigned long long)(__attribute__((address_space(3))) const void*)(p))

constexpr int S_LEN   = 2048;
constexpr int D_MODEL = 1024;
constexpr int NHEAD   = 16;
constexpr int DK      = 64;
constexpr float NEGINF = -1e9f;

// ---------------- fragment loaders ----------------

// A-fragment (16x32, f16) from fp32 row-major [M x 1024], convert on load.
// lane L: row = base + (L&15); K = kstep + 16*(L>>4) + h  (16 contiguous)
__device__ inline v16h afrag_f32(const float* __restrict__ A, int row, int kstep, int lane) {
    const float* p = A + row * D_MODEL + kstep + ((lane >> 4) << 4);
    v16h a;
#pragma unroll
    for (int i = 0; i < 8; ++i) a[i] = (_Float16)p[i];
#pragma unroll
    for (int i = 0; i < 8; ++i) a[8 + i] = (_Float16)p[8 + i];
    return a;
}

// A-fragment from f16 row-major with leading dim ld (Q tiles, P tiles).
__device__ inline v16h afrag_f16(const _Float16* __restrict__ A, int row, int ld, int kstep, int lane) {
    const _Float16* p = A + row * ld + kstep + ((lane >> 4) << 4);
    v16h a;
#pragma unroll
    for (int i = 0; i < 8; ++i) a[i] = p[i];
#pragma unroll
    for (int i = 0; i < 8; ++i) a[8 + i] = p[8 + i];
    return a;
}

// B-fragment (32x16, f16) for y = x * W^T : B[k][n] = W[n][k], fp32 W, convert on load.
// lane L: col = base + (L&15); K = kstep + 16*(L>>4) + h  (16 contiguous)
__device__ inline v16h bfrag_f32(const float* __restrict__ W, int col, int kstep, int lane) {
    const float* p = W + col * D_MODEL + kstep + ((lane >> 4) << 4);
    v16h b;
#pragma unroll
    for (int i = 0; i < 16; ++i) b[i] = (_Float16)p[i];
    return b;
}

// A-fragment gathering the head-split ctx tensor [B,H,S,dk] as logical [B*S, 1024].
__device__ inline v16h afrag_ctx(const _Float16* __restrict__ ctx, int row, int kstep, int lane) {
    int b = row >> 11, s = row & 2047;
    int k0 = kstep + ((lane >> 4) << 4);
    v16h a;
#pragma unroll
    for (int c = 0; c < 2; ++c) {
        int kc = k0 + c * 8;
        int h = kc >> 6, dk = kc & 63;
        const _Float16* p = ctx + (((b << 4) + h) * S_LEN + s) * DK + dk;
#pragma unroll
        for (int i = 0; i < 8; ++i) a[c * 8 + i] = p[i];
    }
    return a;
}

// ---------------- kernel 1: input projection (launched 3x: Q,K,V) ----------------
// grid (N/64=16, M/64=64), block 128 (4 waves). Wave tile 32x32.
// Output head-split f16: [B,H,S,dk]
__global__ __launch_bounds__(128) void proj_kernel(
    const float* __restrict__ A, const float* __restrict__ W,
    _Float16* __restrict__ O) {
    int lane = threadIdx.x & 31;
    int w    = threadIdx.x >> 5;
    int mw = blockIdx.y * 64 + (w >> 1) * 32;
    int nw = blockIdx.x * 64 + (w & 1) * 32;

    v8f acc[2][2] = {{{}, {}}, {{}, {}}};
    for (int ks = 0; ks < D_MODEL; ks += 32) {
        v16h a0 = afrag_f32(A, mw + (lane & 15), ks, lane);
        v16h a1 = afrag_f32(A, mw + 16 + (lane & 15), ks, lane);
        v16h b0 = bfrag_f32(W, nw + (lane & 15), ks, lane);
        v16h b1 = bfrag_f32(W, nw + 16 + (lane & 15), ks, lane);
        acc[0][0] = WMMA_F16(a0, b0, acc[0][0]);
        acc[0][1] = WMMA_F16(a0, b1, acc[0][1]);
        acc[1][0] = WMMA_F16(a1, b0, acc[1][0]);
        acc[1][1] = WMMA_F16(a1, b1, acc[1][1]);
    }
#pragma unroll
    for (int mi = 0; mi < 2; ++mi)
#pragma unroll
        for (int ni = 0; ni < 2; ++ni)
#pragma unroll
            for (int r = 0; r < 8; ++r) {
                int grow = mw + mi * 16 + r + ((lane >> 4) << 3);
                int gcol = nw + ni * 16 + (lane & 15);
                int b = grow >> 11, s = grow & 2047;
                int h = gcol >> 6,  dk = gcol & 63;
                O[(((b << 4) + h) * S_LEN + s) * DK + dk] = (_Float16)acc[mi][ni][r];
            }
}

// ---------------- kernel 2: causal flash attention ----------------
// grid (S/64=32, H=16, B=2), block 128 (4 waves, each 16 query rows).
// K tiles fetched by the Tensor Data Mover; V tiles transposed manually into LDS.
__global__ __launch_bounds__(128) void flash_attn_kernel(
    const _Float16* __restrict__ Qh, const _Float16* __restrict__ Kh,
    const _Float16* __restrict__ Vh, const int* __restrict__ mask,
    _Float16* __restrict__ ctx) {
    __shared__ __align__(32) _Float16 Kt[32 * DK];       // [key][dk]   (TDM target)
    __shared__ __align__(32) _Float16 Vt[DK * 32];       // [dk][key]   (transposed)
    __shared__ __align__(32) _Float16 Pt[4][16 * 32];    // per-wave P tile

    int b = blockIdx.z, h = blockIdx.y, qblk = blockIdx.x;
    int lane = threadIdx.x & 31;
    int w    = threadIdx.x >> 5;
    int qbase = qblk * 64 + w * 16;

    const _Float16* Qp = Qh + (size_t)((b << 4) + h) * S_LEN * DK;
    const _Float16* Kp = Kh + (size_t)((b << 4) + h) * S_LEN * DK;
    const _Float16* Vp = Vh + (size_t)((b << 4) + h) * S_LEN * DK;

    // Q fragments held in registers for the whole kernel (16 x 64 = 2 k-steps)
    v16h qf0 = afrag_f16(Qp, qbase + (lane & 15), DK, 0, lane);
    v16h qf1 = afrag_f16(Qp, qbase + (lane & 15), DK, 32, lane);

    // Loop-invariant half of the TDM descriptor (D#) for the K tile:
    // 2D tensor of f16, tensor_dim0=64 (stride 64), tensor_dim1=2048,
    // tile = 64 x 32 rows -> contiguous [32 keys][64 dk] block into LDS.
    const unsigned kt_lds = LDS_OFFSET(&Kt[0]);
    u32x8 g1;
    g1[0] = 0x00010000u;                    // wg_mask=0, data_size=1 (2 bytes)
    g1[1] = ((unsigned)DK) << 16;           // tensor_dim0 = 64
    g1[2] = ((unsigned)S_LEN) << 16;        // tensor_dim0[31:16]=0 | tensor_dim1 = 2048
    g1[3] = ((unsigned)DK) << 16;           // tensor_dim1[31:16]=0 | tile_dim0 = 64
    g1[4] = 32u;                            // tile_dim1 = 32, tile_dim2 = 0
    g1[5] = (unsigned)DK;                   // tensor_dim0_stride = 64
    g1[6] = 0u;
    g1[7] = 0u;

    float mrow[8], lrow[8];
    v8f o[4] = {{}, {}, {}, {}};
#pragma unroll
    for (int r = 0; r < 8; ++r) { mrow[r] = -3e38f; lrow[r] = 0.f; }

    int nkv = 2 * qblk + 2;  // causal: keys up to qblk*64+63
    for (int kv = 0; kv < nkv; ++kv) {
        int keybase = kv * 32;
        __syncthreads();  // previous iteration done with Kt/Vt/Pt

        // --- K tile via Tensor Data Mover (wave 0 issues; TDM ignores EXEC) ---
        if (w == 0) {
            unsigned long long ga =
                (unsigned long long)(const void*)(Kp + (size_t)keybase * DK);
            u32x4 g0;
            g0[0] = 1u;                                   // count = 1 descriptor
            g0[1] = kt_lds;                               // lds_addr
            g0[2] = (unsigned)ga;                         // global_addr[31:0]
            g0[3] = ((unsigned)(ga >> 32) & 0x01FFFFFFu)  // global_addr[56:32]
                    | (2u << 30);                         // type = 2 (image)
            asm volatile("tensor_load_to_lds %0, %1"
                         :: "s"(g0), "s"(g1) : "memory");
            __builtin_amdgcn_s_wait_tensorcnt(0);
        }

        // --- V tile, transposed into LDS (4 threads per key row) ---
        {
            int key = threadIdx.x >> 2;
            int c   = (threadIdx.x & 3) * 16;
            const _Float16* vs = Vp + (keybase + key) * DK + c;
#pragma unroll
            for (int i = 0; i < 16; ++i) Vt[(c + i) * 32 + key] = vs[i];
        }
        __syncthreads();

        // S = Q * K^T tile (16 queries x 32 keys), two 16-key subtiles
        v8f s0 = {}, s1 = {};
#pragma unroll
        for (int ks = 0; ks < 2; ++ks) {
            int kdk = ks * 32 + ((lane >> 4) << 4);
            v16h bk0 = *(const v16h*)&Kt[((lane & 15)) * DK + kdk];
            v16h bk1 = *(const v16h*)&Kt[(16 + (lane & 15)) * DK + kdk];
            v16h qa = ks ? qf1 : qf0;
            s0 = WMMA_F16(qa, bk0, s0);
            s1 = WMMA_F16(qa, bk1, s1);
        }

        // scale + causal + pad mask, online softmax
        int key0 = keybase + (lane & 15);
        int key1 = key0 + 16;
        int ok0 = mask[b * S_LEN + key0];
        int ok1 = mask[b * S_LEN + key1];
        float al[8];
#pragma unroll
        for (int r = 0; r < 8; ++r) {
            int qg = qbase + r + ((lane >> 4) << 3);
            float v0 = s0[r] * 0.125f;
            float v1 = s1[r] * 0.125f;
            if (key0 > qg || !ok0) v0 = NEGINF;
            if (key1 > qg || !ok1) v1 = NEGINF;
            float t = fmaxf(v0, v1);
#pragma unroll
            for (int m = 1; m <= 8; m <<= 1) t = fmaxf(t, __shfl_xor(t, m, 32));
            float mnew = fmaxf(mrow[r], t);
            float alpha = __expf(mrow[r] - mnew);
            mrow[r] = mnew;
            float p0 = __expf(v0 - mnew);
            float p1 = __expf(v1 - mnew);
            float rs = p0 + p1;
#pragma unroll
            for (int m = 1; m <= 8; m <<= 1) rs += __shfl_xor(rs, m, 32);
            lrow[r] = lrow[r] * alpha + rs;
            al[r] = alpha;
            s0[r] = p0;
            s1[r] = p1;
        }
#pragma unroll
        for (int sub = 0; sub < 4; ++sub)
#pragma unroll
            for (int r = 0; r < 8; ++r) o[sub][r] *= al[r];

        // stage P (C-layout -> LDS -> A-layout)
        _Float16* Pw = &Pt[w][0];
#pragma unroll
        for (int r = 0; r < 8; ++r) {
            int row = r + ((lane >> 4) << 3);
            Pw[row * 32 + (lane & 15)]      = (_Float16)s0[r];
            Pw[row * 32 + 16 + (lane & 15)] = (_Float16)s1[r];
        }
        __syncthreads();  // orders P stores vs A-layout reads (and Kt/Vt reuse)
        v16h pa = afrag_f16(Pw, lane & 15, 32, 0, lane);

        // O += P * V  (K = 32 keys, 4 dk-subtiles)
#pragma unroll
        for (int sub = 0; sub < 4; ++sub) {
            int dk = sub * 16 + (lane & 15);
            v16h bv = *(const v16h*)&Vt[dk * 32 + ((lane >> 4) << 4)];
            o[sub] = WMMA_F16(pa, bv, o[sub]);
        }
    }

    float inv[8];
#pragma unroll
    for (int r = 0; r < 8; ++r) inv[r] = 1.0f / lrow[r];
    _Float16* cp = ctx + (size_t)((b << 4) + h) * S_LEN * DK;
#pragma unroll
    for (int sub = 0; sub < 4; ++sub)
#pragma unroll
        for (int r = 0; r < 8; ++r) {
            int qg = qbase + r + ((lane >> 4) << 3);
            int dk = sub * 16 + (lane & 15);
            cp[qg * DK + dk] = (_Float16)(o[sub][r] * inv[r]);
        }
}

// ---------------- kernel 3: output projection ----------------
// out(f32) = ctx(gathered over heads, f16) @ Wo^T
__global__ __launch_bounds__(128) void out_proj_kernel(
    const _Float16* __restrict__ ctx, const float* __restrict__ Wo,
    float* __restrict__ out) {
    int lane = threadIdx.x & 31;
    int w    = threadIdx.x >> 5;
    int mw = blockIdx.y * 64 + (w >> 1) * 32;
    int nw = blockIdx.x * 64 + (w & 1) * 32;

    v8f acc[2][2] = {{{}, {}}, {{}, {}}};
    for (int ks = 0; ks < D_MODEL; ks += 32) {
        v16h a0 = afrag_ctx(ctx, mw + (lane & 15), ks, lane);
        v16h a1 = afrag_ctx(ctx, mw + 16 + (lane & 15), ks, lane);
        v16h b0 = bfrag_f32(Wo, nw + (lane & 15), ks, lane);
        v16h b1 = bfrag_f32(Wo, nw + 16 + (lane & 15), ks, lane);
        acc[0][0] = WMMA_F16(a0, b0, acc[0][0]);
        acc[0][1] = WMMA_F16(a0, b1, acc[0][1]);
        acc[1][0] = WMMA_F16(a1, b0, acc[1][0]);
        acc[1][1] = WMMA_F16(a1, b1, acc[1][1]);
    }
#pragma unroll
    for (int mi = 0; mi < 2; ++mi)
#pragma unroll
        for (int ni = 0; ni < 2; ++ni)
#pragma unroll
            for (int r = 0; r < 8; ++r) {
                int grow = mw + mi * 16 + r + ((lane >> 4) << 3);
                int gcol = nw + ni * 16 + (lane & 15);
                out[(size_t)grow * D_MODEL + gcol] = acc[mi][ni][r];
            }
}

// ---------------- host launcher ----------------
extern "C" void kernel_launch(void* const* d_in, const int* in_sizes, int n_in,
                              void* d_out, int out_size, void* d_ws, size_t ws_size,
                              hipStream_t stream) {
    const float* q   = (const float*)d_in[0];
    const float* k   = (const float*)d_in[1];
    const float* v   = (const float*)d_in[2];
    const int*  mask = (const int*)  d_in[3];
    const float* Wq  = (const float*)d_in[4];
    const float* Wk  = (const float*)d_in[5];
    const float* Wv  = (const float*)d_in[6];
    const float* Wo  = (const float*)d_in[7];

    constexpr size_t NE = (size_t)2 * NHEAD * S_LEN * DK;  // elems per [B,H,S,dk] tensor
    _Float16* Qh  = (_Float16*)d_ws;
    _Float16* Kh  = Qh + NE;
    _Float16* Vh  = Kh + NE;
    _Float16* ctx = Vh + NE;

    dim3 ggrid(D_MODEL / 64, (2 * S_LEN) / 64);

    // 1) Q/K/V projections: M=4096, N=1024, 64x64 block tiles
    proj_kernel<<<ggrid, 128, 0, stream>>>(q, Wq, Qh);
    proj_kernel<<<ggrid, 128, 0, stream>>>(k, Wk, Kh);
    proj_kernel<<<ggrid, 128, 0, stream>>>(v, Wv, Vh);

    // 2) causal flash attention: (qblocks of 64, heads, batch)
    flash_attn_kernel<<<dim3(S_LEN / 64, NHEAD, 2), 128, 0, stream>>>(
        Qh, Kh, Vh, mask, ctx);

    // 3) output projection -> fp32 d_out
    out_proj_kernel<<<ggrid, 128, 0, stream>>>(ctx, Wo, (float*)d_out);
}
